// DilatedAttention2_53429393162513
// MI455X (gfx1250) — compile-verified, matrix operand order checked by reading, not run on hardware
//
#include <hip/hip_runtime.h>

// ---------------------------------------------------------------------------
// Types for CDNA5 WMMA (wave32): 16x16x32 f16 -> f32
// ---------------------------------------------------------------------------
typedef __attribute__((ext_vector_type(16))) _Float16 v16h;
typedef __attribute__((ext_vector_type(8)))  _Float16 v8h;
typedef __attribute__((ext_vector_type(8)))  float    v8f;
typedef __attribute__((ext_vector_type(4)))  int      i32x4;

#define WMMA_F16(a, b, c) \
  __builtin_amdgcn_wmma_f32_16x16x32_f16(false, (a), false, (b), (short)0, (c), false, false)

// Per-lane fragment load for 16-bit 16x32 A (and mirrored B) layout:
// lane = 16*half + idx ; VGPR pairs 0..3 hold K = 8*half + {0..7},
// VGPR pairs 4..7 hold K = 16 + 8*half + {0..7}.  Caller passes
// p = base + row*ld + k0 + 8*half ; chunks at p and p+16 (each 16B aligned).
__device__ __forceinline__ v16h load_frag(const _Float16* p) {
  union { v16h v; v8h h[2]; } u;
  u.h[0] = *(const v8h*)(p);
  u.h[1] = *(const v8h*)(p + 16);
  return u.v;
}

// ---------------------------------------------------------------------------
// Async global -> LDS staging (CDNA5 ASYNCcnt path) with portable fallback.
// Builtin expects int4-vector pointers: (global i32x4*, shared i32x4*, imm, cpol)
// ---------------------------------------------------------------------------
#if __has_builtin(__builtin_amdgcn_global_load_async_to_lds_b128) && \
    __has_builtin(__builtin_amdgcn_s_wait_asynccnt)
#define HAVE_ASYNC_LDS 1
__device__ __forceinline__ void stage16B(const _Float16* g, _Float16* l) {
  __builtin_amdgcn_global_load_async_to_lds_b128(
      (__attribute__((address_space(1))) i32x4*)(g),
      (__attribute__((address_space(3))) i32x4*)(l), 0, 0);
}
#define WAIT_ASYNC() __builtin_amdgcn_s_wait_asynccnt(0)
#else
#define HAVE_ASYNC_LDS 0
__device__ __forceinline__ void stage16B(const _Float16* g, _Float16* l) {
  *(v8h*)l = *(const v8h*)g;   // global_load_b128 + ds_store_b128
}
#define WAIT_ASYNC() ((void)0)
#endif

static constexpr int D = 1024;   // model dim == head dim
static constexpr int S = 256;    // tokens per segment after dilation

// ---------------------------------------------------------------------------
// 1) Dilated gather x -> xs (f16).  xs row r = (b,g,t) <- x row b*4096+g*512+2t
// ---------------------------------------------------------------------------
__global__ __launch_bounds__(256) void gather_xs(const float* __restrict__ x,
                                                 _Float16* __restrict__ xs) {
  size_t i = ((size_t)blockIdx.x * 256 + threadIdx.x) * 4;   // 4 elems/thread
  int r = (int)(i >> 10);
  int d = (int)(i & 1023);
  int b = r >> 11, rem = r & 2047, g = rem >> 8, t = rem & 255;
  const float* src = x + (((size_t)(b * 4096 + g * 512 + 2 * t)) << 10) + d;
  _Float16* dst = xs + i;
  for (int j = 0; j < 4; ++j) dst[j] = (_Float16)src[j];
}

// 1M-element f32 -> f16 conversion (weights)
__global__ __launch_bounds__(256) void conv_w(const float* __restrict__ s,
                                              _Float16* __restrict__ d) {
  size_t i = ((size_t)blockIdx.x * 256 + threadIdx.x) * 4;
  float4 v = *(const float4*)(s + i);
  d[i + 0] = (_Float16)v.x;  d[i + 1] = (_Float16)v.y;
  d[i + 2] = (_Float16)v.z;  d[i + 3] = (_Float16)v.w;
}

// ---------------------------------------------------------------------------
// 2) WMMA GEMM core: block = 8 waves = 128(M) x 64(N) tile, K-step 32.
//    B tile (64 rows x 32 k, f16, 4KB) double-buffered in LDS, staged
//    asynchronously; A fragments double-buffered in registers.
// ---------------------------------------------------------------------------
struct GemmAcc { v8f a0, a1, a2, a3; };

template <typename StageBuf>
__device__ __forceinline__ GemmAcc gemm_core(const _Float16* __restrict__ A,
                                             const _Float16* __restrict__ W,
                                             StageBuf& sB,      // [2][64][32]
                                             int mBase, int nBase) {
  const int tid  = threadIdx.x;
  const int lane = tid & 31;
  const int ln   = lane & 15;
  const int half = lane >> 4;

  // staging assignment: thread -> (row n in tile, 16B chunk)
  const int sn = tid >> 2;          // 0..63
  const int sc = (tid & 3) * 8;     // halfs: 0,8,16,24
  const _Float16* wSrc = W + (size_t)(nBase + sn) * D + sc;   // B[k,n] = W[n,k]
  const _Float16* aP   = A + (size_t)(mBase + ln) * D + 8 * half;

  // stage k=0 into buffer 0
  stage16B(wSrc, &sB[0][sn][sc]);
  WAIT_ASYNC();
  __syncthreads();

  GemmAcc g;
  g.a0 = (v8f){}; g.a1 = (v8f){}; g.a2 = (v8f){}; g.a3 = (v8f){};
  v16h aCur = load_frag(aP);
  for (int k0 = 0; k0 < D; k0 += 32) {
    const int cur = (k0 >> 5) & 1;
    const int nxt = cur ^ 1;
    v16h aNxt;
    const bool more = (k0 + 32) < D;
    if (more) {
      stage16B(wSrc + k0 + 32, &sB[nxt][sn][sc]);   // overlap with compute
      __builtin_prefetch(aP + k0 + 64, 0, 0);
      aNxt = load_frag(aP + k0 + 32);
    }
    v16h b0 = load_frag(&sB[cur][ln +  0][8 * half]);   // ds_load_b128 x2 each
    v16h b1 = load_frag(&sB[cur][ln + 16][8 * half]);
    v16h b2 = load_frag(&sB[cur][ln + 32][8 * half]);
    v16h b3 = load_frag(&sB[cur][ln + 48][8 * half]);
    g.a0 = WMMA_F16(aCur, b0, g.a0);
    g.a1 = WMMA_F16(aCur, b1, g.a1);
    g.a2 = WMMA_F16(aCur, b2, g.a2);
    g.a3 = WMMA_F16(aCur, b3, g.a3);
    if (more) {
      aCur = aNxt;
      WAIT_ASYNC();
      __syncthreads();
    }
  }
  return g;
}

// C[8192,1024](f16) = A[8192,1024] @ W[1024,1024]^T
__global__ __launch_bounds__(256) void gemm_xw16(const _Float16* __restrict__ A,
                                                 const _Float16* __restrict__ W,
                                                 _Float16* __restrict__ C) {
  __shared__ __align__(16) _Float16 sB[2][64][32];
  const int lane = threadIdx.x & 31;
  const int wid  = threadIdx.x >> 5;
  const int ln   = lane & 15;
  const int half = lane >> 4;
  const int mBase = blockIdx.x * 128 + wid * 16;
  const int nBase = blockIdx.y * 64;

  GemmAcc g = gemm_core(A, W, sB, mBase, nBase);

  // C/D layout: VGPR r -> M = r + 8*half, N = lane%16
  _Float16* cP = C + (size_t)mBase * D + nBase + ln;
  for (int r = 0; r < 8; ++r) {
    const size_t row = (size_t)(r + 8 * half) * D;
    cP[row +  0] = (_Float16)g.a0[r];
    cP[row + 16] = (_Float16)g.a1[r];
    cP[row + 32] = (_Float16)g.a2[r];
    cP[row + 48] = (_Float16)g.a3[r];
  }
}

// out[b, 2m, :] = Y[b*2048+m, :] @ Wo^T   (f32 output, even-row scatter)
__global__ __launch_bounds__(256) void gemm_out16(const _Float16* __restrict__ Yh,
                                                  const _Float16* __restrict__ Wo,
                                                  float* __restrict__ out) {
  __shared__ __align__(16) _Float16 sB[2][64][32];
  const int lane = threadIdx.x & 31;
  const int wid  = threadIdx.x >> 5;
  const int ln   = lane & 15;
  const int half = lane >> 4;
  const int mBase = blockIdx.x * 128 + wid * 16;
  const int nBase = blockIdx.y * 64;

  GemmAcc g = gemm_core(Yh, Wo, sB, mBase, nBase);

  for (int r = 0; r < 8; ++r) {
    const int rg = mBase + r + 8 * half;      // 0..8191
    const int b = rg >> 11, m = rg & 2047;
    float* oP = out + (((size_t)(b * 4096 + 2 * m)) << 10) + nBase + ln;
    oP[ 0] = g.a0[r];
    oP[16] = g.a1[r];
    oP[32] = g.a2[r];
    oP[48] = g.a3[r];
  }
}

// ---------------------------------------------------------------------------
// 3) In-place LayerNorm over 24576 contiguous rows (q,k,v).  Rows >= 16384
//    belong to V and are additionally written transposed into Vt[bg][d][t].
// ---------------------------------------------------------------------------
__global__ __launch_bounds__(256) void ln_kernel(_Float16* __restrict__ qkv,
                                                 _Float16* __restrict__ vt,
                                                 const float* __restrict__ gamma,
                                                 const float* __restrict__ beta) {
  const int row = blockIdx.x;
  _Float16* p = qkv + (size_t)row * D;
  const int tid = threadIdx.x;

  float vals[4], s = 0.f, s2 = 0.f;
  for (int j = 0; j < 4; ++j) {
    float xv = (float)p[tid + 256 * j];
    vals[j] = xv;  s += xv;  s2 += xv * xv;
  }
  for (int off = 16; off; off >>= 1) {
    s  += __shfl_xor(s,  off, 32);
    s2 += __shfl_xor(s2, off, 32);
  }
  __shared__ float red[20];
  const int wid = tid >> 5, lane = tid & 31;
  if (lane == 0) { red[wid] = s; red[8 + wid] = s2; }
  __syncthreads();
  if (tid == 0) {
    float a = 0.f, b = 0.f;
    for (int w = 0; w < 8; ++w) { a += red[w]; b += red[8 + w]; }
    red[16] = a; red[17] = b;
  }
  __syncthreads();
  const float mean = red[16] * (1.0f / 1024.0f);
  const float var  = red[17] * (1.0f / 1024.0f) - mean * mean;
  const float inv  = rsqrtf(var + 1e-5f);

  const bool isV = row >= 16384;
  const int vrow = row - 16384;
  const int bg = vrow >> 8, t = vrow & 255;
  for (int j = 0; j < 4; ++j) {
    const int d = tid + 256 * j;
    float y = (vals[j] - mean) * inv * gamma[d] + beta[d];
    _Float16 yh = (_Float16)y;
    p[d] = yh;
    if (isV) vt[(size_t)bg * (D * S) + (size_t)d * S + t] = yh;
  }
}

// ---------------------------------------------------------------------------
// 4) Attention per (bg, 16-query block):  S = (Q Kt)*scale, causal softmax,
//    Y = attn V.  Both matmul phases use WMMA; attn A-frags come from LDS.
// ---------------------------------------------------------------------------
__global__ __launch_bounds__(256) void attn_kernel(const _Float16* __restrict__ Q,
                                                   const _Float16* __restrict__ K,
                                                   const _Float16* __restrict__ Vt,
                                                   _Float16* __restrict__ Y) {
  __shared__ __align__(16) float    sS[16][256];   // scores (f32)
  __shared__ __align__(16) _Float16 sA[16][256];   // softmaxed attn (f16)

  const int bg = blockIdx.x >> 4;          // 0..31
  const int qb = blockIdx.x & 15;          // 16-query block within segment
  const _Float16* Qb  = Q  + (size_t)bg * S * D;
  const _Float16* Kb  = K  + (size_t)bg * S * D;
  const _Float16* Vtb = Vt + (size_t)bg * D * S;

  const int lane = threadIdx.x & 31;
  const int wid  = threadIdx.x >> 5;
  const int ln   = lane & 15;
  const int half = lane >> 4;
  const int mBase = qb * 16;

  // ---- phase 1: scores, wave w owns key columns [w*32, w*32+32) ----
  {
    const _Float16* aP  = Qb + (size_t)(mBase + ln) * D;
    const _Float16* b0P = Kb + (size_t)(wid * 32 + ln) * D;
    const _Float16* b1P = Kb + (size_t)(wid * 32 + 16 + ln) * D;
    v8f acc0 = {}, acc1 = {};
    for (int d0 = 0; d0 < D; d0 += 32) {
      const int ko = d0 + 8 * half;
      v16h a  = load_frag(aP + ko);
      v16h b0 = load_frag(b0P + ko);
      v16h b1 = load_frag(b1P + ko);
      acc0 = WMMA_F16(a, b0, acc0);
      acc1 = WMMA_F16(a, b1, acc1);
    }
    const float scale = 0.03125f;            // 1/sqrt(1024)
    for (int r = 0; r < 8; ++r) {
      const int qi = r + 8 * half;
      const int qg = mBase + qi;             // query index within segment
      const int k0 = wid * 32 + ln;
      const int k1 = wid * 32 + 16 + ln;
      sS[qi][k0] = (k0 <= qg) ? acc0[r] * scale : -1e30f;
      sS[qi][k1] = (k1 <= qg) ? acc1[r] * scale : -1e30f;
    }
  }
  __syncthreads();

  // ---- phase 2: softmax, wave w owns rows {2w, 2w+1} ----
  for (int rr = 0; rr < 2; ++rr) {
    const int qi = wid * 2 + rr;
    float vb[8], mx = -1e30f;
    for (int j = 0; j < 8; ++j) { vb[j] = sS[qi][lane + 32 * j]; mx = fmaxf(mx, vb[j]); }
    for (int off = 16; off; off >>= 1) mx = fmaxf(mx, __shfl_xor(mx, off, 32));
    float sum = 0.f;
    for (int j = 0; j < 8; ++j) { float e = __expf(vb[j] - mx); vb[j] = e; sum += e; }
    for (int off = 16; off; off >>= 1) sum += __shfl_xor(sum, off, 32);
    const float inv = 1.0f / sum;
    for (int j = 0; j < 8; ++j) sA[qi][lane + 32 * j] = (_Float16)(vb[j] * inv);
  }
  __syncthreads();

  // ---- phase 3: Y strip, wave w owns d in [w*128, w*128+128) ----
  v8f acc[8];
  for (int t = 0; t < 8; ++t) acc[t] = (v8f){};
  for (int k0 = 0; k0 < S; k0 += 32) {
    v16h a = load_frag(&sA[ln][k0 + 8 * half]);           // ds_load_b128 x2
    for (int t = 0; t < 8; ++t) {
      const int dcol = wid * 128 + t * 16 + ln;
      v16h b = load_frag(Vtb + (size_t)dcol * S + k0 + 8 * half);
      acc[t] = WMMA_F16(a, b, acc[t]);
    }
  }
  _Float16* Yb = Y + (size_t)bg * S * D;
  for (int t = 0; t < 8; ++t)
    for (int r = 0; r < 8; ++r) {
      const int qi = r + 8 * half;
      Yb[(size_t)(mBase + qi) * D + wid * 128 + t * 16 + ln] = (_Float16)acc[t][r];
    }
}

// ---------------------------------------------------------------------------
// 5) Zero odd output rows (dilated scatter leaves them zero)
// ---------------------------------------------------------------------------
__global__ __launch_bounds__(256) void zero_odd(float* __restrict__ out) {
  size_t i = ((size_t)blockIdx.x * 256 + threadIdx.x) * 4;  // index in odd space
  int r = (int)(i >> 10);
  int d = (int)(i & 1023);
  int b = r >> 11, m = r & 2047;
  float4 z = {0.f, 0.f, 0.f, 0.f};
  *(float4*)(out + (((size_t)(b * 4096 + 2 * m + 1)) << 10) + d) = z;
}

// ---------------------------------------------------------------------------
// Host launcher
// ---------------------------------------------------------------------------
extern "C" void kernel_launch(void* const* d_in, const int* in_sizes, int n_in,
                              void* d_out, int out_size, void* d_ws, size_t ws_size,
                              hipStream_t stream) {
  (void)in_sizes; (void)n_in; (void)out_size; (void)ws_size;
  const float* x     = (const float*)d_in[0];
  const float* Wq    = (const float*)d_in[1];
  const float* Wk    = (const float*)d_in[2];
  const float* Wv    = (const float*)d_in[3];
  const float* Wo    = (const float*)d_in[4];
  const float* gamma = (const float*)d_in[5];
  const float* beta  = (const float*)d_in[6];
  float* out = (float*)d_out;

  _Float16* ws = (_Float16*)d_ws;
  const size_t M1 = (size_t)1 << 20;           // 1M halfs
  _Float16* xs = ws;                           // 8M  (reused as Vt later)
  _Float16* wq = ws + 8  * M1;                 // 1M each
  _Float16* wk = ws + 9  * M1;
  _Float16* wv = ws + 10 * M1;
  _Float16* wo = ws + 11 * M1;
  _Float16* qh = ws + 12 * M1;                 // 8M (q,k,v contiguous for LN)
  _Float16* kh = ws + 20 * M1;
  _Float16* vh = ws + 28 * M1;                 // reused as Y after attention
  _Float16* vt = xs;                           // [bg][1024][256]
  _Float16* yh = vh;
  // total: 36M halfs = 72 MB workspace

  gather_xs<<<8192, 256, 0, stream>>>(x, xs);
  conv_w<<<1024, 256, 0, stream>>>(Wq, wq);
  conv_w<<<1024, 256, 0, stream>>>(Wk, wk);
  conv_w<<<1024, 256, 0, stream>>>(Wv, wv);
  conv_w<<<1024, 256, 0, stream>>>(Wo, wo);

  dim3 gemmGrid(64, 16);                       // 128x64 tile per block
  gemm_xw16<<<gemmGrid, 256, 0, stream>>>(xs, wq, qh);
  gemm_xw16<<<gemmGrid, 256, 0, stream>>>(xs, wk, kh);
  gemm_xw16<<<gemmGrid, 256, 0, stream>>>(xs, wv, vh);

  ln_kernel<<<24576, 256, 0, stream>>>(qh, vt, gamma, beta);

  attn_kernel<<<512, 256, 0, stream>>>(qh, kh, vt, yh);

  zero_odd<<<8192, 256, 0, stream>>>(out);
  gemm_out16<<<gemmGrid, 256, 0, stream>>>(yh, wo, out);
}
